// VNN_DGCNN_4535485464855
// MI455X (gfx1250) — matrix operand order repaired and checked by reading, not compiled, not run on hardware
//
#include <hip/hip_runtime.h>
#include <cstddef>

// ---------------------------------------------------------------------------
// VN-DGCNN encoder for MI455X (gfx1250, wave32, WMMA).
// All GEMM work (KNN pairwise inner products, per-edge VN linears, concat VN
// linear, gpool linear) runs on v_wmma_f32_16x16x32_f16 (f16 in, f32 accum).
// Pointwise/reduction stages are bandwidth-shaped with channel-contiguous
// (B,N,3,C) layouts. BN statistics use a deterministic 2-stage reduction.
// ---------------------------------------------------------------------------

typedef __attribute__((ext_vector_type(16))) _Float16 v16h;
typedef __attribute__((ext_vector_type(8)))  _Float16 v8h;
typedef __attribute__((ext_vector_type(8)))  float    v8f;
typedef __attribute__((ext_vector_type(4)))  float    v4f;

#define DEV __device__ __forceinline__

constexpr int Bb = 4;      // batch
constexpr int Nn = 2048;   // points
constexpr int Kn = 10;     // knn
constexpr int Hh = 128;    // hidden / c_dim
constexpr float EPSv   = 1e-6f;
constexpr float BNEPS  = 1e-5f;
constexpr float SLOPEv = 0.2f;

DEV v16h comb16(v8h lo, v8h hi){
  v16h r;
#pragma unroll
  for(int i=0;i<8;++i){ r[i]=lo[i]; r[i+8]=hi[i]; }
  return r;
}

// A-matrix fragment (16x32 f16, M x K), tile row-major [16][ld] in LDS.
// lane l(0-15): M=l, halves K = g*8..g*8+7 and 16+g*8..16+g*8+7 (g = lane>>4)
DEV v16h frag_a_lds(const _Float16* t, int ld, int ks){
  int lane = threadIdx.x & 31;
  const _Float16* p = t + (lane & 15)*ld + ks*32 + (lane >> 4)*8;
  return comb16(*(const v8h*)p, *(const v8h*)(p + 16));
}
// B-matrix fragment (32x16 f16, K x N), tile column-major [16][ld] in LDS.
// lane l(0-15): N=l, halves K = g*8..g*8+7 and 16+g*8..16+g*8+7 (g = lane>>4)
DEV v16h frag_b_lds(const _Float16* t, int ld, int ks){
  int lane = threadIdx.x & 31;
  const _Float16* p = t + (lane & 15)*ld + ks*32 + (lane >> 4)*8;
  return comb16(*(const v8h*)p, *(const v8h*)(p + 16));
}
// Weight A-fragment straight from global (row-major [rows][ld], L2-hot).
DEV v16h frag_w_glb(const _Float16* W, int ld, int m0, int ks){
  int lane = threadIdx.x & 31;
  const _Float16* p = W + (size_t)(m0 + (lane & 15))*ld + ks*32 + (lane >> 4)*8;
  return comb16(*(const v8h*)p, *(const v8h*)(p + 16));
}

DEV v8f wmma16(v16h a, v16h b, v8f c){
  return __builtin_amdgcn_wmma_f32_16x16x32_f16(false, a, false, b, (short)0, c, false, false);
}

// ---------------------------------------------------------------------------
// Weight f32 -> padded f16 (zero pad columns to Kp)
// ---------------------------------------------------------------------------
__global__ void k_cvt_w(const float* __restrict__ src, _Float16* __restrict__ dst,
                        int rows, int cols, int ldp){
  for(int i = blockIdx.x*blockDim.x + threadIdx.x; i < rows*ldp; i += gridDim.x*blockDim.x){
    int r = i / ldp, c = i % ldp;
    dst[i] = (c < cols) ? (_Float16)src[(size_t)r*cols + c] : (_Float16)0.f;
  }
}

// ---------------------------------------------------------------------------
// Squared norms per point, FT layout (B,N,3,C) -> contiguous 3C per point
// ---------------------------------------------------------------------------
__global__ void k_xx(const float* __restrict__ FT, float* __restrict__ xx, int D3){
  int pt = blockIdx.x*blockDim.x + threadIdx.x;
  if(pt >= Bb*Nn) return;
  const float* p = FT + (size_t)pt * D3;
  float s = 0.f;
  for(int d=0; d<D3; ++d){ float v = p[d]; s += v*v; }
  xx[pt] = s;
}

// ---------------------------------------------------------------------------
// KNN top-10 via WMMA inner products.
// Ranking key: 2*<xi,xj> - ||xj||^2 (the -||xi||^2 term is row-constant).
// Block = 4 waves; each wave owns one 16-query tile (B operand, resident in
// LDS); candidate 16-point tiles (A operand) staged cooperatively & shared.
// ---------------------------------------------------------------------------
template<int DPAD>
__global__ void __launch_bounds__(128) k_knn(const float* __restrict__ FT,
                                             const float* __restrict__ xx,
                                             int* __restrict__ idxout, int D)
{
  __shared__ __align__(16) _Float16 sQ[4][16*DPAD];  // query tiles, col-major per point
  __shared__ __align__(16) _Float16 sC[16*DPAD];     // candidate tile, row-major
  __shared__ float sXX[16];
  int tid = threadIdx.x, wave = tid >> 5, lane = tid & 31;
  int b  = blockIdx.x / (Nn/64);
  int q0 = (blockIdx.x % (Nn/64)) * 64;

  { // stage this wave's 16 query rows (f32 -> f16, zero-pad to DPAD)
    const float* src = FT + ((size_t)b*Nn + q0 + wave*16) * (size_t)D;
    _Float16* dst = sQ[wave];
    for(int i = lane; i < 16*DPAD; i += 32){
      int qr = i / DPAD, d = i % DPAD;
      dst[i] = (d < D) ? (_Float16)src[(size_t)qr*D + d] : (_Float16)0.f;
    }
  }
  float bd[Kn]; int bi[Kn];
#pragma unroll
  for(int t=0;t<Kn;++t){ bd[t] = -3.0e38f; bi[t] = 0; }
  __syncthreads();

  for(int ct = 0; ct < Nn/16; ++ct){
    { // cooperative candidate tile staging
      const float* src = FT + ((size_t)b*Nn + ct*16) * (size_t)D;
      for(int i = tid; i < 16*DPAD; i += 128){
        int cr = i / DPAD, d = i % DPAD;
        sC[i] = (d < D) ? (_Float16)src[(size_t)cr*D + d] : (_Float16)0.f;
      }
      if(tid < 16) sXX[tid] = xx[(size_t)b*Nn + ct*16 + tid];
    }
    __syncthreads();
    v8f acc = {};
#pragma unroll
    for(int ks = 0; ks < DPAD/32; ++ks)
      acc = wmma16(frag_a_lds(sC, DPAD, ks), frag_b_lds(sQ[wave], DPAD, ks), acc);
    // D layout: lane (g=lane>>4, n=lane&15) component r -> (M = g*8+r, N = n)
    int g = lane >> 4;
#pragma unroll
    for(int r = 0; r < 8; ++r){
      float sc = 2.f*acc[r] - sXX[g*8 + r];
      int j = ct*16 + g*8 + r;
      if(sc > bd[Kn-1]){
        float v = sc; int ix = j;
#pragma unroll
        for(int t = 0; t < Kn; ++t){
          bool gt = v > bd[t];
          float tv = bd[t]; int ti = bi[t];
          bd[t] = gt ? v : tv;  bi[t] = gt ? ix : ti;
          v = gt ? tv : v;      ix = gt ? ti : ix;
        }
      }
    }
    __syncthreads();
  }
  // merge lane l (cols g=0) with lane l+16 (cols g=1)
  float pv[Kn]; int pi[Kn];
#pragma unroll
  for(int t=0;t<Kn;++t){
    pv[t] = __shfl(bd[t], (lane + 16) & 31, 32);
    pi[t] = __shfl(bi[t], (lane + 16) & 31, 32);
  }
  if(lane < 16){
#pragma unroll
    for(int t=0;t<Kn;++t){
      float v = pv[t]; int ix = pi[t];
      if(v > bd[Kn-1]){
#pragma unroll
        for(int u=0;u<Kn;++u){
          bool gt = v > bd[u];
          float tv = bd[u]; int ti = bi[u];
          bd[u] = gt ? v : tv;  bi[u] = gt ? ix : ti;
          v = gt ? tv : v;      ix = gt ? ti : ix;
        }
      }
    }
    int q = q0 + wave*16 + lane;
    int* op = idxout + ((size_t)b*Nn + q)*Kn;
#pragma unroll
    for(int t=0;t<Kn;++t) op[t] = bi[t];
  }
}

// ---------------------------------------------------------------------------
// Fused gather + dual GEMM:  P = Wf*E, D = Wd*E.
// MODE 0 (edge): columns = (b,n,k,axis); E = [feat_j - feat_n ; feat_n] (2C)
// MODE 1 (plain): columns = (b,n,axis);  E = FT row (C values)
// 8 waves = 8 M-tiles (128 output channels). Output layout: col*128 + ch
// (channel fastest -> each lane stores two contiguous float4).
// KP / MODE / HASD are compile-time: fully unrolled straight-line WMMA pairs,
// no per-kstep scalar branching, exactly-sized LDS tile.
// ---------------------------------------------------------------------------
template<int KP, int MODE, bool HASD>
__global__ void __launch_bounds__(256) k_pair_gemm(
    const float* __restrict__ FT, const int* __restrict__ idx,
    const _Float16* __restrict__ Wf, const _Float16* __restrict__ Wd,
    float* __restrict__ P, float* __restrict__ Dout,
    int C, int ncols, int tiles_per_wg)
{
  __shared__ __align__(16) _Float16 sE[16*KP];   // E tile, col-major [16][KP]
  int tid = threadIdx.x, wave = tid >> 5, lane = tid & 31;
  int m0 = wave * 16;
  for(int t = 0; t < tiles_per_wg; ++t){
    int tile = blockIdx.x*tiles_per_wg + t;
    if(tile*16 >= ncols) break;
    int colbase = tile*16;
    { // stage E tile: 16 threads per column, contiguous channel reads
      int csub = tid & 15, colo = tid >> 4;
      int col = colbase + colo;
      _Float16* dcol = sE + colo*KP;
      if(MODE == 0){
        int e = col/3, axis = col - e*3;
        int kk = e % Kn;
        int n  = (e / Kn) % Nn;
        int b  = e / (Kn*Nn);
        int j  = idx[((size_t)b*Nn + n)*Kn + kk];
        const float* pn = FT + (((size_t)b*Nn + n)*3 + axis)*(size_t)C;
        const float* pj = FT + (((size_t)b*Nn + j)*3 + axis)*(size_t)C;
        for(int c = csub; c < C; c += 16){
          float fn = pn[c], fj = pj[c];
          dcol[c]     = (_Float16)(fj - fn);
          dcol[C + c] = (_Float16)fn;
        }
        for(int c = 2*C + csub; c < KP; c += 16) dcol[c] = (_Float16)0.f;
      } else {
        const float* pc = FT + (size_t)col*(size_t)C;
        for(int c = csub; c < C; c += 16) dcol[c] = (_Float16)pc[c];
        for(int c = C + csub; c < KP; c += 16) dcol[c] = (_Float16)0.f;
      }
    }
    __syncthreads();
    v8f accP = {}, accD = {};
#pragma unroll
    for(int ks = 0; ks < KP/32; ++ks){
      v16h bf = frag_b_lds(sE, KP, ks);
      accP = wmma16(frag_w_glb(Wf, KP, m0, ks), bf, accP);
      if(HASD) accD = wmma16(frag_w_glb(Wd, KP, m0, ks), bf, accD);
    }
    {
      int g = lane >> 4, n16 = lane & 15;
      size_t col = (size_t)colbase + n16;
      float* pp = P + col*Hh + m0 + g*8;
      v4f lo, hi;
#pragma unroll
      for(int r=0;r<4;++r){ lo[r]=accP[r]; hi[r]=accP[r+4]; }
      *(v4f*)pp = lo; *(v4f*)(pp+4) = hi;
      if(HASD){
        float* pd = Dout + col*Hh + m0 + g*8;
#pragma unroll
        for(int r=0;r<4;++r){ lo[r]=accD[r]; hi[r]=accD[r+4]; }
        *(v4f*)pd = lo; *(v4f*)(pd+4) = hi;
      }
    }
    __syncthreads();
  }
}

// ---------------------------------------------------------------------------
// Per-channel BN stats of ||p|| over elements; deterministic 2-stage reduce.
// P layout: (elements, 3, 128). partial: [block][256] = {sum(128), sumsq(128)}
// ---------------------------------------------------------------------------
__global__ void __launch_bounds__(256) k_bnstats(const float* __restrict__ P,
                                                 float* __restrict__ partial, int nE)
{
  __shared__ float s1[256], s2[256];
  int tid = threadIdx.x;
  int ch = tid & (Hh-1);
  int half = tid >> 7;
  float a1 = 0.f, a2 = 0.f;
  for(int e = blockIdx.x*2 + half; e < nE; e += gridDim.x*2){
    const float* p = P + (size_t)e*3*Hh + ch;
    float x0 = p[0], x1 = p[Hh], x2 = p[2*Hh];
    float nr = sqrtf(x0*x0 + x1*x1 + x2*x2) + EPSv;
    a1 += nr; a2 += nr*nr;
  }
  s1[tid] = a1; s2[tid] = a2;
  __syncthreads();
  if(tid < Hh){
    partial[(size_t)blockIdx.x*256 + tid]        = s1[tid] + s1[tid+128];
    partial[(size_t)blockIdx.x*256 + 128 + tid]  = s2[tid] + s2[tid+128];
  }
}

__global__ void k_redstats(const float* __restrict__ partial, float* __restrict__ stats, int nblocks){
  int tid = threadIdx.x;     // 256 threads: [0,128)=sum, [128,256)=sumsq
  float a = 0.f;
  for(int b = 0; b < nblocks; ++b) a += partial[(size_t)b*256 + tid];
  stats[tid] = a;
}

// ---------------------------------------------------------------------------
// Finalize edge layer: BN scale + vector leaky relu + mean over K.
// Writes next-layer FT (B,N,3,128) and optionally a 128-wide slice of
// xcT (B,N,3,512).
// ---------------------------------------------------------------------------
__global__ void __launch_bounds__(256) k_fin_edge(
    const float* __restrict__ P, const float* __restrict__ Dm,
    const float* __restrict__ stats, const float* __restrict__ gamma,
    const float* __restrict__ beta, float* __restrict__ FTout,
    float* __restrict__ xcT, int choff, float invcnt)
{
  int t = blockIdx.x*blockDim.x + threadIdx.x;
  if(t >= Bb*Nn*Hh) return;
  int ch = t & (Hh-1);
  int pt = t >> 7;
  float mean = stats[ch]*invcnt;
  float var  = stats[Hh+ch]*invcnt - mean*mean;
  float rstd = rsqrtf(var + BNEPS);
  float ga = gamma[ch], be = beta[ch];
  float a0=0.f, a1=0.f, a2=0.f;
  for(int k = 0; k < Kn; ++k){
    size_t e = (size_t)pt*Kn + k;
    const float* pp = P  + e*(3*Hh) + ch;
    const float* pd = Dm + e*(3*Hh) + ch;
    float p0=pp[0], p1=pp[Hh], p2=pp[2*Hh];
    float d0=pd[0], d1=pd[Hh], d2=pd[2*Hh];
    float nr  = sqrtf(p0*p0 + p1*p1 + p2*p2) + EPSv;
    float nbn = (nr - mean)*rstd*ga + be;
    float sc  = nbn / nr;
    p0 *= sc; p1 *= sc; p2 *= sc;
    float dot = p0*d0 + p1*d1 + p2*d2;
    float dns = d0*d0 + d1*d1 + d2*d2 + EPSv;
    float w = (dot < 0.f) ? ((1.f - SLOPEv)*dot/dns) : 0.f;
    a0 += p0 - w*d0; a1 += p1 - w*d1; a2 += p2 - w*d2;
  }
  float inv = 1.f/(float)Kn;
  a0 *= inv; a1 *= inv; a2 *= inv;
  float* o = FTout + (size_t)pt*3*Hh + ch;
  o[0] = a0; o[Hh] = a1; o[2*Hh] = a2;
  if(xcT){
    float* o2 = xcT + (size_t)pt*3*512 + choff + ch;
    o2[0] = a0; o2[512] = a1; o2[1024] = a2;
  }
}

// Finalize concat layer (no neighbor mean): writes xc2T (B,N,3,128).
__global__ void __launch_bounds__(256) k_fin_plain(
    const float* __restrict__ P, const float* __restrict__ Dm,
    const float* __restrict__ stats, const float* __restrict__ gamma,
    const float* __restrict__ beta, float* __restrict__ outT, float invcnt)
{
  int t = blockIdx.x*blockDim.x + threadIdx.x;
  if(t >= Bb*Nn*Hh) return;
  int ch = t & (Hh-1);
  int pt = t >> 7;
  float mean = stats[ch]*invcnt;
  float var  = stats[Hh+ch]*invcnt - mean*mean;
  float rstd = rsqrtf(var + BNEPS);
  float ga = gamma[ch], be = beta[ch];
  const float* pp = P  + (size_t)pt*(3*Hh) + ch;
  const float* pd = Dm + (size_t)pt*(3*Hh) + ch;
  float p0=pp[0], p1=pp[Hh], p2=pp[2*Hh];
  float d0=pd[0], d1=pd[Hh], d2=pd[2*Hh];
  float nr  = sqrtf(p0*p0 + p1*p1 + p2*p2) + EPSv;
  float nbn = (nr - mean)*rstd*ga + be;
  float sc  = nbn / nr;
  p0 *= sc; p1 *= sc; p2 *= sc;
  float dot = p0*d0 + p1*d1 + p2*d2;
  float dns = d0*d0 + d1*d1 + d2*d2 + EPSv;
  float w = (dot < 0.f) ? ((1.f - SLOPEv)*dot/dns) : 0.f;
  float* o = outT + (size_t)pt*3*Hh + ch;
  o[0] = p0 - w*d0; o[Hh] = p1 - w*d1; o[2*Hh] = p2 - w*d2;
}

// ---------------------------------------------------------------------------
// Global pool: block per (b, ch). argmax_n of <xc, d> over axis, gather vector
// at argmax. Tie-break = first occurrence (matches jnp.argmax).
// ---------------------------------------------------------------------------
__global__ void __launch_bounds__(256) k_gpool_argmax(const float* __restrict__ xc2T,
                                                      const float* __restrict__ G,
                                                      float* __restrict__ out)
{
  int b  = blockIdx.x >> 7;
  int ch = blockIdx.x & 127;
  __shared__ float sbest[256];
  __shared__ int   sidx[256];
  float best = -3.0e38f; int bidx = 0;
  for(int n = threadIdx.x; n < Nn; n += 256){
    size_t base = ((size_t)(b*Nn + n)*3)*Hh + ch;
    float dot = xc2T[base]*G[base] + xc2T[base+Hh]*G[base+Hh] + xc2T[base+2*Hh]*G[base+2*Hh];
    if(dot > best){ best = dot; bidx = n; }
  }
  sbest[threadIdx.x] = best; sidx[threadIdx.x] = bidx;
  __syncthreads();
  for(int s = 128; s > 0; s >>= 1){
    if(threadIdx.x < s){
      bool take = (sbest[threadIdx.x+s] > sbest[threadIdx.x]) ||
                  (sbest[threadIdx.x+s] == sbest[threadIdx.x] && sidx[threadIdx.x+s] < sidx[threadIdx.x]);
      if(take){ sbest[threadIdx.x] = sbest[threadIdx.x+s]; sidx[threadIdx.x] = sidx[threadIdx.x+s]; }
    }
    __syncthreads();
  }
  if(threadIdx.x == 0){
    int n = sidx[0];
    size_t base = ((size_t)(b*Nn + n)*3)*Hh + ch;
    float* o = out + ((size_t)b*Hh + ch)*3;
    o[0] = xc2T[base]; o[1] = xc2T[base+Hh]; o[2] = xc2T[base+2*Hh];
  }
}

// ---------------------------------------------------------------------------
extern "C" void kernel_launch(void* const* d_in, const int* in_sizes, int n_in,
                              void* d_out, int out_size, void* d_ws, size_t ws_size,
                              hipStream_t stream)
{
  (void)in_sizes; (void)n_in; (void)out_size; (void)ws_size;
  const float* x      = (const float*)d_in[0];
  const float* Wf_pos = (const float*)d_in[1];
  const float* Wd_pos = (const float*)d_in[2];
  const float* g_pos  = (const float*)d_in[3];
  const float* b_pos  = (const float*)d_in[4];
  const float* Wf_l[4] = {(const float*)d_in[5],(const float*)d_in[9],(const float*)d_in[13],(const float*)d_in[17]};
  const float* Wd_l[4] = {(const float*)d_in[6],(const float*)d_in[10],(const float*)d_in[14],(const float*)d_in[18]};
  const float* g_l[4]  = {(const float*)d_in[7],(const float*)d_in[11],(const float*)d_in[15],(const float*)d_in[19]};
  const float* b_l[4]  = {(const float*)d_in[8],(const float*)d_in[12],(const float*)d_in[16],(const float*)d_in[20]};
  const float* Wf_c = (const float*)d_in[21];
  const float* Wd_c = (const float*)d_in[22];
  const float* g_c  = (const float*)d_in[23];
  const float* b_c  = (const float*)d_in[24];
  const float* W_gp = (const float*)d_in[25];

  char* base = (char*)d_ws;
  size_t off = 0;
  auto alloc = [&](size_t bytes)->char*{
    char* p = base + off;
    off = (off + bytes + 255) & ~(size_t)255;
    return p;
  };

  _Float16* wfp = (_Float16*)alloc((size_t)128*32*2);
  _Float16* wdp = (_Float16*)alloc((size_t)128*32*2);
  _Float16* wf16[4]; _Float16* wd16[4];
  for(int l=0;l<4;++l){ wf16[l]=(_Float16*)alloc((size_t)128*256*2); wd16[l]=(_Float16*)alloc((size_t)128*256*2); }
  _Float16* wfc = (_Float16*)alloc((size_t)128*512*2);
  _Float16* wdc = (_Float16*)alloc((size_t)128*512*2);
  _Float16* wg  = (_Float16*)alloc((size_t)128*128*2);
  float* xxb    = (float*)alloc((size_t)Bb*Nn*4);
  int*   idxb   = (int*)  alloc((size_t)Bb*Nn*Kn*4);
  float* statsb = (float*)alloc((size_t)256*4);
  float* partb  = (float*)alloc((size_t)512*256*4);
  size_t ftbytes = (size_t)Bb*Nn*3*Hh*4;
  float* FTh    = (float*)alloc(ftbytes);
  float* FTx[4];
  for(int l=0;l<4;++l) FTx[l] = (float*)alloc(ftbytes);
  float* xcT    = (float*)alloc((size_t)Bb*Nn*3*512*4);
  float* xc2T   = (float*)alloc(ftbytes);
  float* Gb     = (float*)alloc(ftbytes);
  float* Pb     = (float*)alloc((size_t)Bb*Nn*Kn*3*Hh*4);
  float* Db     = (float*)alloc((size_t)Bb*Nn*Kn*3*Hh*4);

  // --- weight conversion (f32 -> padded f16) ---
  k_cvt_w<<<64,256,0,stream>>>(Wf_pos, wfp, 128,   2,  32);
  k_cvt_w<<<64,256,0,stream>>>(Wd_pos, wdp, 128,   2,  32);
  for(int l=0;l<4;++l){
    k_cvt_w<<<64,256,0,stream>>>(Wf_l[l], wf16[l], 128, 256, 256);
    k_cvt_w<<<64,256,0,stream>>>(Wd_l[l], wd16[l], 128, 256, 256);
  }
  k_cvt_w<<<64,256,0,stream>>>(Wf_c, wfc, 128, 512, 512);
  k_cvt_w<<<64,256,0,stream>>>(Wd_c, wdc, 128, 512, 512);
  k_cvt_w<<<64,256,0,stream>>>(W_gp, wg,  128, 128, 128);

  const int nE_edge = Bb*Nn*Kn;            // 81920
  const int nc_edge = Bb*Nn*Kn*3;          // 245760 columns
  const int nc_pln  = Bb*Nn*3;             // 24576 columns
  const int fin_grid = (Bb*Nn*Hh)/256;     // 4096

  auto edge_layer = [&](const float* FTin, int C, const _Float16* wf, const _Float16* wd,
                        const float* gamma, const float* beta, float* FTout, float* xcTs, int choff){
    int D3 = 3*C;
    k_xx<<<(Bb*Nn+255)/256, 256, 0, stream>>>(FTin, xxb, D3);
    if(C == 1){
      k_knn<32> <<<Bb*Nn/64, 128, 0, stream>>>(FTin, xxb, idxb, D3);
      k_pair_gemm<32,0,true><<<nc_edge/64, 256, 0, stream>>>(FTin, idxb, wf, wd, Pb, Db, C, nc_edge, 4);
    } else {
      k_knn<384><<<Bb*Nn/64, 128, 0, stream>>>(FTin, xxb, idxb, D3);
      k_pair_gemm<256,0,true><<<nc_edge/64, 256, 0, stream>>>(FTin, idxb, wf, wd, Pb, Db, C, nc_edge, 4);
    }
    k_bnstats<<<512, 256, 0, stream>>>(Pb, partb, nE_edge);
    k_redstats<<<1, 256, 0, stream>>>(partb, statsb, 512);
    k_fin_edge<<<fin_grid, 256, 0, stream>>>(Pb, Db, statsb, gamma, beta, FTout, xcTs, choff,
                                             1.f/(float)nE_edge);
  };

  // pos layer (input x is already (B,N,3) == (B,N,3,C=1) transposed layout)
  edge_layer(x,      1,   wfp,     wdp,     g_pos,  b_pos,  FTh,    nullptr, 0);
  // EdgeConv layers 0..3, writing both the next FT and the xcT concat slice
  edge_layer(FTh,    128, wf16[0], wd16[0], g_l[0], b_l[0], FTx[0], xcT,   0);
  edge_layer(FTx[0], 128, wf16[1], wd16[1], g_l[1], b_l[1], FTx[1], xcT, 128);
  edge_layer(FTx[1], 128, wf16[2], wd16[2], g_l[2], b_l[2], FTx[2], xcT, 256);
  edge_layer(FTx[2], 128, wf16[3], wd16[3], g_l[3], b_l[3], FTx[3], xcT, 384);

  // concat layer (plain VN linear on 512 channels)
  k_pair_gemm<512,1,true><<<nc_pln/64, 256, 0, stream>>>(xcT, nullptr, wfc, wdc, Pb, Db, 512, nc_pln, 4);
  k_bnstats<<<512, 256, 0, stream>>>(Pb, partb, Bb*Nn);
  k_redstats<<<1, 256, 0, stream>>>(partb, statsb, 512);
  k_fin_plain<<<fin_grid, 256, 0, stream>>>(Pb, Db, statsb, g_c, b_c, xc2T, 1.f/(float)(Bb*Nn));

  // gpool direction GEMM + argmax gather
  k_pair_gemm<128,1,false><<<nc_pln/64, 256, 0, stream>>>(xc2T, nullptr, wg, nullptr, Gb, nullptr, 128, nc_pln, 4);
  k_gpool_argmax<<<Bb*Hh, 256, 0, stream>>>(xc2T, Gb, (float*)d_out);
}
